// P1GCN_18330920419815
// MI455X (gfx1250) — matrix operand order repaired and checked by reading, not compiled, not run on hardware
//
#include <hip/hip_runtime.h>

typedef __bf16 bf16;
typedef __attribute__((ext_vector_type(16))) __bf16 v16bf;
typedef __attribute__((ext_vector_type(8)))  __bf16 v8bf;
typedef __attribute__((ext_vector_type(8)))  float  v8f;

#define N_NODES 100000
#define N_EDGES 1600000
#define MTILES  (N_NODES / 16)   // 6250, exact

// ---------------------------------------------------------------------------
// fragment loaders (wave32, contiguous k order, same in A and B)
// ---------------------------------------------------------------------------

// A fragment (16x32 bf16) from LDS, row-major [16][ld].
// lane L: m = L&15, half = L>>4.  e in 0..15 -> K = k0+16*half+e (contiguous).
// Two 16B DS loads.
__device__ __forceinline__ v16bf load_a_frag(const bf16* As, int ld, int k0, int lane) {
  const int m = lane & 15, half = lane >> 4;
  const bf16* p = As + m * ld + k0 + 16 * half;
  v8bf lo = *(const v8bf*)p;
  v8bf hi = *(const v8bf*)(p + 8);
  return __builtin_shufflevector(lo, hi, 0,1,2,3,4,5,6,7,8,9,10,11,12,13,14,15);
}

// B fragment (32x16 bf16) from global, weights pre-transposed to [N][K] bf16.
// lane L: n = L&15, half = L>>4.  e in 0..15 -> K = k0 + 16*half + e (contiguous).
__device__ __forceinline__ v16bf load_b_frag(const bf16* WT, int K, int k0, int lane) {
  const int n = lane & 15, half = lane >> 4;
  return *(const v16bf*)(WT + (size_t)n * K + k0 + 16 * half);
}

__device__ __forceinline__ void atomAddF(float* p, float v) {
  __hip_atomic_fetch_add(p, v, __ATOMIC_RELAXED, __HIP_MEMORY_SCOPE_AGENT);
}

// ---------------------------------------------------------------------------
// utility kernels
// ---------------------------------------------------------------------------
__global__ void k_zero4(float4* __restrict__ p, int n4) {
  int i = blockIdx.x * blockDim.x + threadIdx.x;
  if (i < n4) p[i] = make_float4(0.f, 0.f, 0.f, 0.f);
}

// W [K][N] f32  ->  WT [N][K] bf16
__global__ void k_cvtT(const float* __restrict__ W, bf16* __restrict__ WT, int K, int N) {
  int i = blockIdx.x * blockDim.x + threadIdx.x;
  if (i < K * N) {
    int k = i / N, n = i - k * N;
    WT[(size_t)n * K + k] = (bf16)W[i];
  }
}

__global__ void k_deg(const int* __restrict__ dst, float* __restrict__ deg, int nE) {
  int i = blockIdx.x * blockDim.x + threadIdx.x;
  if (i < nE) atomAddF(&deg[dst[i]], 1.0f);
}

__global__ void k_invdeg(float* __restrict__ deg, int n) {
  int i = blockIdx.x * blockDim.x + threadIdx.x;
  if (i < n) deg[i] = 1.0f / fmaxf(deg[i], 1.0f);
}

// segment-sum scatter: agg[dst] += h[src], feature dim D (vectorized by 4)
template<int D>
__global__ void k_scatter(const int* __restrict__ src, const int* __restrict__ dst,
                          const float* __restrict__ h, float* __restrict__ agg, int nE) {
  constexpr int DV = D / 4;
  int tid = blockIdx.x * blockDim.x + threadIdx.x;
  int e = tid / DV;
  if (e >= nE) return;
  int f4 = (tid - e * DV) * 4;
  int s = src[e], d = dst[e];
  float4 v = *(const float4*)(h + (size_t)s * D + f4);
  float* p = agg + (size_t)d * D + f4;
  atomAddF(p + 0, v.x); atomAddF(p + 1, v.y);
  atomAddF(p + 2, v.z); atomAddF(p + 3, v.w);
}

// ---------------------------------------------------------------------------
// WMMA GEMM:  out[16 rows/block, NOUT] = relu( (A * rowscale?) @ W + bias )
// A: [N_NODES][K] f32, WT: [NOUT][K] bf16, one wave per 16x16 tile.
// ---------------------------------------------------------------------------
template<int K, int NOUT, bool SCALED>
__global__ __launch_bounds__(2 * NOUT) void k_gemm16(
    const float* __restrict__ A, const float* __restrict__ rowscale,
    const bf16* __restrict__ WT, const float* __restrict__ bias,
    float* __restrict__ out) {
  constexpr int LD = K + 16;                 // pad to dodge DS bank conflicts
  __shared__ alignas(32) bf16 As[16 * LD];
  const int row0 = blockIdx.x * 16;

  for (int i = threadIdx.x; i < 16 * K; i += 2 * NOUT) {
    int r = i / K, c = i - r * K;
    float v = A[(size_t)(row0 + r) * K + c];
    if (SCALED) v *= rowscale[row0 + r];
    As[r * LD + c] = (bf16)v;
  }
  __syncthreads();

  const int lane = threadIdx.x & 31;
  const int ncol0 = (threadIdx.x >> 5) << 4;
  v8f acc = {};
#pragma unroll
  for (int k0 = 0; k0 < K; k0 += 32) {
    v16bf a = load_a_frag(As, LD, k0, lane);
    v16bf b = load_b_frag(WT + (size_t)ncol0 * K, K, k0, lane);
    acc = __builtin_amdgcn_wmma_f32_16x16x32_bf16(false, a, false, b, (short)0, acc, false, false);
  }
  const int n = lane & 15, half = lane >> 4;
  const float bv = bias[ncol0 + n];
#pragma unroll
  for (int r = 0; r < 8; ++r) {
    float v = fmaxf(acc[r] + bv, 0.0f);
    out[(size_t)(row0 + r + 8 * half) * NOUT + ncol0 + n] = v;
  }
}

// ---------------------------------------------------------------------------
// batchnorm stats: per-thread fixed-feature partial sums (stride multiple of 64)
// ---------------------------------------------------------------------------
__global__ void k_bnstat(const float* __restrict__ H, float* __restrict__ sums,
                         float* __restrict__ sumsq) {
  const long total = (long)N_NODES * 64;
  const int stride = gridDim.x * blockDim.x;          // 1024*256 = 262144, 64 | stride
  float s = 0.f, q = 0.f;
  for (long i = (long)blockIdx.x * blockDim.x + threadIdx.x; i < total; i += stride) {
    float v = H[i]; s += v; q += v * v;
  }
  int f = threadIdx.x & 63;                            // == global_tid & 63
  atomAddF(&sums[f], s);
  atomAddF(&sumsq[f], q);
}

__global__ void k_bnfinal(const float* __restrict__ sums, const float* __restrict__ sumsq,
                          const float* __restrict__ gamma, const float* __restrict__ beta,
                          float* __restrict__ scale, float* __restrict__ shift) {
  int f = threadIdx.x;
  if (f < 64) {
    float mean = sums[f] * (1.0f / (float)N_NODES);
    float var  = sumsq[f] * (1.0f / (float)N_NODES) - mean * mean;
    float sc   = gamma[f] * rsqrtf(var + 1e-5f);
    scale[f] = sc;
    shift[f] = beta[f] - mean * sc;
  }
}

// ---------------------------------------------------------------------------
// fused lin2+lin3: out = relu((Hbn) @ W2 + b2) @ W3 + b3, Hbn = H*scale+shift
// block = 16 rows, 2 waves (N=32). h2 tile bounced through LDS between WMMAs.
// ---------------------------------------------------------------------------
__global__ __launch_bounds__(64) void k_lin23(
    const float* __restrict__ H, const float* __restrict__ scale, const float* __restrict__ shift,
    const bf16* __restrict__ W2T, const float* __restrict__ b2,
    const bf16* __restrict__ W3T, const float* __restrict__ b3,
    float* __restrict__ out) {
  __shared__ alignas(32) bf16 As[16 * 80];   // 16 x (64+16)
  __shared__ alignas(32) bf16 Hs[16 * 48];   // 16 x (32+16)
  const int row0 = blockIdx.x * 16;

  for (int i = threadIdx.x; i < 16 * 64; i += 64) {
    int r = i >> 6, c = i & 63;
    As[r * 80 + c] = (bf16)(H[(size_t)(row0 + r) * 64 + c] * scale[c] + shift[c]);
  }
  __syncthreads();

  const int lane = threadIdx.x & 31;
  const int ncol0 = (threadIdx.x >> 5) << 4;
  const int n = lane & 15, half = lane >> 4;

  v8f acc = {};
#pragma unroll
  for (int k0 = 0; k0 < 64; k0 += 32) {
    v16bf a = load_a_frag(As, 80, k0, lane);
    v16bf b = load_b_frag(W2T + (size_t)ncol0 * 64, 64, k0, lane);
    acc = __builtin_amdgcn_wmma_f32_16x16x32_bf16(false, a, false, b, (short)0, acc, false, false);
  }
  float bv = b2[ncol0 + n];
#pragma unroll
  for (int r = 0; r < 8; ++r)
    Hs[(r + 8 * half) * 48 + ncol0 + n] = (bf16)fmaxf(acc[r] + bv, 0.0f);
  __syncthreads();

  v8f acc2 = {};
  {
    v16bf a = load_a_frag(Hs, 48, 0, lane);
    v16bf b = load_b_frag(W3T + (size_t)ncol0 * 32, 32, 0, lane);
    acc2 = __builtin_amdgcn_wmma_f32_16x16x32_bf16(false, a, false, b, (short)0, acc2, false, false);
  }
  float b3v = b3[ncol0 + n];
#pragma unroll
  for (int r = 0; r < 8; ++r)
    out[(size_t)(row0 + r + 8 * half) * 32 + ncol0 + n] = acc2[r] + b3v;
}

// ---------------------------------------------------------------------------
// host orchestration
// ---------------------------------------------------------------------------
extern "C" void kernel_launch(void* const* d_in, const int* in_sizes, int n_in,
                              void* d_out, int out_size, void* d_ws, size_t ws_size,
                              hipStream_t stream) {
  const float* x    = (const float*)d_in[0];
  const float* W1   = (const float*)d_in[1];
  const float* b1   = (const float*)d_in[2];
  const float* Wc1  = (const float*)d_in[3];
  const float* bc1  = (const float*)d_in[4];
  const float* Wc2  = (const float*)d_in[5];
  const float* bc2  = (const float*)d_in[6];
  const float* Wc3  = (const float*)d_in[7];
  const float* bc3  = (const float*)d_in[8];
  const float* gmm  = (const float*)d_in[9];
  const float* beta = (const float*)d_in[10];
  const float* W2   = (const float*)d_in[11];
  const float* b2   = (const float*)d_in[12];
  const float* W3   = (const float*)d_in[13];
  const float* b3   = (const float*)d_in[14];
  const int*  edges = (const int*)d_in[15];
  const int* srcI = edges;
  const int* dstI = edges + N_EDGES;
  float* out = (float*)d_out;

  char* wp = (char*)d_ws;
  auto carve = [&](size_t bytes) -> void* {
    void* p = (void*)wp; wp += (bytes + 255) & ~(size_t)255; return p;
  };
  bf16*  W1T   = (bf16*)carve((size_t)512 * 128 * sizeof(bf16));
  bf16*  Wc1T  = (bf16*)carve((size_t)128 * 64 * sizeof(bf16));
  bf16*  Wc2T  = (bf16*)carve((size_t)64 * 64 * sizeof(bf16));
  bf16*  Wc3T  = (bf16*)carve((size_t)64 * 64 * sizeof(bf16));
  bf16*  W2T   = (bf16*)carve((size_t)64 * 32 * sizeof(bf16));
  bf16*  W3T   = (bf16*)carve((size_t)32 * 32 * sizeof(bf16));
  float* deg   = (float*)carve((size_t)N_NODES * sizeof(float));
  float* stats = (float*)carve(256 * sizeof(float)); // sums|sumsq|scale|shift
  float* bufA  = (float*)carve((size_t)N_NODES * 128 * sizeof(float));
  float* bufB  = (float*)carve((size_t)N_NODES * 128 * sizeof(float));
  float* bufC  = (float*)carve((size_t)N_NODES * 64 * sizeof(float));

  auto blocks = [](long n, int b) { return (unsigned)((n + b - 1) / b); };

  // weights -> transposed bf16
  k_cvtT<<<blocks(512 * 128, 256), 256, 0, stream>>>(W1,  W1T,  512, 128);
  k_cvtT<<<blocks(128 * 64, 256), 256, 0, stream>>>(Wc1, Wc1T, 128, 64);
  k_cvtT<<<blocks(64 * 64, 256), 256, 0, stream>>>(Wc2, Wc2T, 64, 64);
  k_cvtT<<<blocks(64 * 64, 256), 256, 0, stream>>>(Wc3, Wc3T, 64, 64);
  k_cvtT<<<blocks(64 * 32, 256), 256, 0, stream>>>(W2,  W2T,  64, 32);
  k_cvtT<<<blocks(32 * 32, 256), 256, 0, stream>>>(W3,  W3T,  32, 32);

  // degrees -> 1/max(deg,1)
  k_zero4<<<blocks(N_NODES / 4, 256), 256, 0, stream>>>((float4*)deg, N_NODES / 4);
  k_deg<<<blocks(N_EDGES, 256), 256, 0, stream>>>(dstI, deg, N_EDGES);
  k_invdeg<<<blocks(N_NODES, 256), 256, 0, stream>>>(deg, N_NODES);

  // lin1: h1 = relu(x @ W1 + b1)   [N,128]
  k_gemm16<512, 128, false><<<MTILES, 256, 0, stream>>>(x, nullptr, W1T, b1, bufA);

  // conv1
  k_zero4<<<blocks((long)N_NODES * 32, 256), 256, 0, stream>>>((float4*)bufB, N_NODES * 32);
  k_scatter<128><<<blocks((long)N_EDGES * 32, 256), 256, 0, stream>>>(srcI, dstI, bufA, bufB, N_EDGES);
  k_gemm16<128, 64, true><<<MTILES, 128, 0, stream>>>(bufB, deg, Wc1T, bc1, bufC);

  // conv2
  k_zero4<<<blocks((long)N_NODES * 16, 256), 256, 0, stream>>>((float4*)bufB, N_NODES * 16);
  k_scatter<64><<<blocks((long)N_EDGES * 16, 256), 256, 0, stream>>>(srcI, dstI, bufC, bufB, N_EDGES);
  k_gemm16<64, 64, true><<<MTILES, 128, 0, stream>>>(bufB, deg, Wc2T, bc2, bufA);

  // conv3
  k_zero4<<<blocks((long)N_NODES * 16, 256), 256, 0, stream>>>((float4*)bufB, N_NODES * 16);
  k_scatter<64><<<blocks((long)N_EDGES * 16, 256), 256, 0, stream>>>(srcI, dstI, bufA, bufB, N_EDGES);
  k_gemm16<64, 64, true><<<MTILES, 128, 0, stream>>>(bufB, deg, Wc3T, bc3, bufC);

  // batchnorm stats + affine fold
  k_zero4<<<1, 64, 0, stream>>>((float4*)stats, 64);
  k_bnstat<<<1024, 256, 0, stream>>>(bufC, stats, stats + 64);
  k_bnfinal<<<1, 64, 0, stream>>>(stats, stats + 64, gmm, beta, stats + 128, stats + 192);

  // fused lin2 + lin3 -> out [N,32]
  k_lin23<<<MTILES, 64, 0, stream>>>(bufC, stats + 128, stats + 192,
                                     W2T, b2, W3T, b3, out);
}
